// TransformerBlock_84645215470163
// MI455X (gfx1250) — matrix operand order repaired and checked by reading, not compiled, not run on hardware
//
#include <hip/hip_runtime.h>
#include <math.h>

#define DM   1024          // D_MODEL
#define NH   16            // NUM_HEADS
#define HD   64            // head dim
#define DFF  4096          // D_FF
#define SEQ  2048
#define BATCH 2
#define TOK  (BATCH * SEQ) // 4096 tokens
#define EPSF 1e-5f

typedef __attribute__((ext_vector_type(16))) _Float16 v16h;
typedef __attribute__((ext_vector_type(8)))  _Float16 v8h;
typedef __attribute__((ext_vector_type(8)))  float    v8f;
typedef __attribute__((ext_vector_type(4)))  unsigned int u32x4;
typedef __attribute__((ext_vector_type(8)))  int      i32x8;
typedef __attribute__((ext_vector_type(4)))  int      i32x4;

#if __has_builtin(__builtin_amdgcn_tensor_load_to_lds)
#define USE_TDM 1
#else
#define USE_TDM 0
#endif

// ---------------------------------------------------------------------------
// WMMA helpers (CDNA5 wave32, 16x16x32 f16 -> f32)
// ---------------------------------------------------------------------------
__device__ __forceinline__ v8f wmma_f16(v16h a, v16h b, v8f c) {
  return __builtin_amdgcn_wmma_f32_16x16x32_f16(false, a, false, b, (short)0, c,
                                                false, false);
}

__device__ __forceinline__ v16h pack16(v8h lo, v8h hi) {
  v16h r;
#pragma unroll
  for (int i = 0; i < 8; ++i) { r[i] = lo[i]; r[8 + i] = hi[i]; }
  return r;
}

// A fragment: 16x32 f16 tile from row-major [rows, ld], base at (row0, k0).
__device__ __forceinline__ v16h load_a_frag(const _Float16* base, int ld) {
  const int lane = threadIdx.x & 31;
  const _Float16* p = base + (lane & 15) * ld + ((lane >> 4) << 3);
  return pack16(*(const v8h*)p, *(const v8h*)(p + 16));
}

// B fragment: 32x16 tile of B = Bt^T, Bt row-major [N, ld], base at (n0, k0).
__device__ __forceinline__ v16h load_b_frag(const _Float16* base, int ld) {
  const int lane = threadIdx.x & 31;
  const _Float16* p = base + (lane & 15) * ld + ((lane >> 4) << 3);
  return pack16(*(const v8h*)p, *(const v8h*)(p + 16));
}

// ---------------------------------------------------------------------------
// TDM: 2-D f16 tile load global -> LDS (D# per CDNA5 ISA §8.3/8.4).
// tile_d0 elements contiguous (dim0), tile_d1 rows, row stride = ld elements.
// Uniform (block-scalar) arguments; issued once per calling wave.
// ---------------------------------------------------------------------------
#if USE_TDM
__device__ __forceinline__ void tdm_load_2d(unsigned int lds_off, const void* gaddr,
                                            unsigned int tile_d0, unsigned int tile_d1,
                                            unsigned long long stride_elems) {
  unsigned long long ga = (unsigned long long)gaddr;
  u32x4 g0;
  g0[0] = 1u;                                        // count=1, user descriptor
  g0[1] = lds_off;                                   // lds_addr (bytes)
  g0[2] = (unsigned int)ga;                          // global_addr[31:0]
  g0[3] = (unsigned int)((ga >> 32) & 0x1FFFFFFu)    // global_addr[56:32]
          | (2u << 30);                              // type=2 ("image")
  const unsigned int td0 = 0x40000000u;              // huge tensor dims:
  const unsigned int td1 = 0x40000000u;              //   disable OOB clamping
  i32x8 g1;
  g1[0] = (int)(1u << 16);                           // data_size=1 (2 bytes)
  g1[1] = (int)((td0 & 0xFFFFu) << 16);              // tensor_dim0[15:0]
  g1[2] = (int)((td0 >> 16) | ((td1 & 0xFFFFu) << 16));
  g1[3] = (int)((td1 >> 16) | (tile_d0 << 16));      // tile_dim0
  g1[4] = (int)(tile_d1 & 0xFFFFu);                  // tile_dim1 (tile_dim2=0)
  g1[5] = (int)(unsigned int)stride_elems;           // tensor_dim0_stride lo
  g1[6] = (int)(unsigned int)((stride_elems >> 32) & 0xFFFFu);
  g1[7] = 0;
  i32x4 z4 = {0, 0, 0, 0};
  i32x8 z8 = {0, 0, 0, 0, 0, 0, 0, 0};
  // 6-arg form (clang-23 / therock-10.0 headers): groups 0..3 + extra + cpol
  __builtin_amdgcn_tensor_load_to_lds(g0, g1, z4, z4, z8, 0);
}
#endif

// ---------------------------------------------------------------------------
// fp32 -> f16 cast (weights)
// ---------------------------------------------------------------------------
__global__ void k_cast_f16(const float* __restrict__ in, _Float16* __restrict__ out, int n) {
  int i = blockIdx.x * blockDim.x + threadIdx.x;
  if (i < n) out[i] = (_Float16)in[i];
}

// ---------------------------------------------------------------------------
// RMSNorm: fp32 in, f16 out. One 256-thread block per token row.
// ---------------------------------------------------------------------------
__global__ __launch_bounds__(256)
void k_rmsnorm(const float* __restrict__ x, const float* __restrict__ g,
               _Float16* __restrict__ out) {
  const int row = blockIdx.x;
  const float* xr = x + (size_t)row * DM;
  float ss = 0.f;
  for (int i = threadIdx.x; i < DM; i += 256) { float v = xr[i]; ss += v * v; }
#pragma unroll
  for (int m = 16; m; m >>= 1) ss += __shfl_xor(ss, m, 32);
  __shared__ float red[8];
  __shared__ float s_inv;
  if ((threadIdx.x & 31) == 0) red[threadIdx.x >> 5] = ss;
  __syncthreads();
  if (threadIdx.x == 0) {
    float t = 0.f;
#pragma unroll
    for (int i = 0; i < 8; ++i) t += red[i];
    s_inv = rsqrtf(t / (float)DM + EPSF);
  }
  __syncthreads();
  const float inv = s_inv;
  _Float16* o = out + (size_t)row * DM;
  for (int i = threadIdx.x; i < DM; i += 256)
    o[i] = (_Float16)(xr[i] * inv * g[i]);
}

// ---------------------------------------------------------------------------
// Generic WMMA GEMM: C[M,N] = A[M,K] * B[N,K]^T (+epilogue)
// block = 128 threads (4 waves); wave tile = 64x64; block tile = 64x256.
// A tile (64x32 per K-step, shared by all 4 waves) staged in LDS, double-
// buffered via TDM tensor_load_to_lds (fallback: cooperative vector copy).
// B fragments stream straight from global (L2-resident weights).
// ---------------------------------------------------------------------------
enum { EPI_F16 = 0, EPI_VT = 1, EPI_F32_RES = 2 };

template <int EPI>
__global__ __launch_bounds__(128)
void k_gemm(const _Float16* __restrict__ A, const _Float16* __restrict__ B,
            int M, int N, int K, void* __restrict__ Cout,
            const float* __restrict__ residual) {
  __shared__ __align__(16) _Float16 sA[2][64 * 32];
  const int wave = threadIdx.x >> 5;
  const int lane = threadIdx.x & 31;
  const int m0 = blockIdx.x * 64;
  const int n0 = blockIdx.y * 256 + wave * 64;
  (void)M;

  v8f acc[4][4] = {};
  const _Float16* Ab = A + (size_t)m0 * K;
  const _Float16* Bb = B + (size_t)n0 * K;
  const int nK = K / 32;

#if USE_TDM
  if (threadIdx.x < 32)
    tdm_load_2d((unsigned int)(uintptr_t)&sA[0][0], Ab, 32, 64, (unsigned long long)K);
#endif

  for (int it = 0; it < nK; ++it) {
    const int k0 = it * 32;
    const int cur = it & 1;
#if USE_TDM
    if (threadIdx.x < 32) {
      if (it + 1 < nK) {
        tdm_load_2d((unsigned int)(uintptr_t)&sA[cur ^ 1][0], Ab + k0 + 32, 32, 64,
                    (unsigned long long)K);
        __builtin_amdgcn_s_wait_tensorcnt(1);   // current tile complete
      } else {
        __builtin_amdgcn_s_wait_tensorcnt(0);
      }
    }
    __syncthreads();
#else
    // cooperative staging: 64 rows x 64B, 128 threads x 2 chunks of 16B
    for (int c = threadIdx.x; c < 256; c += 128) {
      const int row = c >> 2;
      const int col8 = (c & 3) << 3;
      *(v8h*)&sA[cur][row * 32 + col8] = *(const v8h*)&Ab[(size_t)row * K + k0 + col8];
    }
    __syncthreads();
#endif

    v16h a[4];
#pragma unroll
    for (int i = 0; i < 4; ++i)
      a[i] = load_a_frag(&sA[cur][i * 16 * 32], 32);
#pragma unroll
    for (int j = 0; j < 4; ++j) {
      if (it + 1 < nK) __builtin_prefetch(Bb + (size_t)j * 16 * K + k0 + 32, 0, 1);
      v16h b = load_b_frag(Bb + (size_t)j * 16 * K + k0, K);
#pragma unroll
      for (int i = 0; i < 4; ++i)
        acc[i][j] = wmma_f16(a[i], b, acc[i][j]);
    }
    __syncthreads();
  }

  // C layout: VGPR r, lane<16 -> M=r (N=lane), lane>=16 -> M=r+8
  const int hi = (lane >> 4) << 3;
  const int nc = lane & 15;
#pragma unroll
  for (int i = 0; i < 4; ++i)
#pragma unroll
    for (int j = 0; j < 4; ++j)
#pragma unroll
      for (int r = 0; r < 8; ++r) {
        const int Mr = m0 + i * 16 + r + hi;
        const int Nc = n0 + j * 16 + nc;
        const float v = acc[i][j][r];
        if (EPI == EPI_F16) {
          ((_Float16*)Cout)[(size_t)Mr * N + Nc] = (_Float16)v;
        } else if (EPI == EPI_VT) {
          // V stored transposed per head: vt[b][h][d][s], s contiguous
          const int h = Nc >> 6, d = Nc & (HD - 1);
          const int bb = Mr >> 11, sl = Mr & (SEQ - 1);
          ((_Float16*)Cout)[(((size_t)(bb * NH + h) * HD + d) << 11) + sl] = (_Float16)v;
        } else {
          ((float*)Cout)[(size_t)Mr * N + Nc] = v + residual[(size_t)Mr * N + Nc];
        }
      }
}

// ---------------------------------------------------------------------------
// Dual-B WMMA GEMM with fused SwiGLU gate: out = silu(A*W1^T) * (A*W3^T), f16
// wave tile 32x32 (x2 accumulators); block = 4 waves -> 32x128 block tile.
// ---------------------------------------------------------------------------
__global__ __launch_bounds__(128)
void k_gemm_gate(const _Float16* __restrict__ A, const _Float16* __restrict__ B1,
                 const _Float16* __restrict__ B3, int M, int N, int K,
                 _Float16* __restrict__ Gout) {
  const int wave = threadIdx.x >> 5;
  const int lane = threadIdx.x & 31;
  const int m0 = blockIdx.x * 32;
  const int n0 = blockIdx.y * 128 + wave * 32;
  (void)M;

  v8f acc1[2][2] = {};
  v8f acc3[2][2] = {};
  const _Float16* Ab = A + (size_t)m0 * K;

  for (int k0 = 0; k0 < K; k0 += 32) {
    v16h a0 = load_a_frag(Ab + k0, K);
    v16h a1 = load_a_frag(Ab + (size_t)16 * K + k0, K);
#pragma unroll
    for (int j = 0; j < 2; ++j) {
      const size_t boff = (size_t)(n0 + j * 16) * K + k0;
      v16h b1 = load_b_frag(B1 + boff, K);
      v16h b3 = load_b_frag(B3 + boff, K);
      acc1[0][j] = wmma_f16(a0, b1, acc1[0][j]);
      acc1[1][j] = wmma_f16(a1, b1, acc1[1][j]);
      acc3[0][j] = wmma_f16(a0, b3, acc3[0][j]);
      acc3[1][j] = wmma_f16(a1, b3, acc3[1][j]);
    }
  }

  const int hi = (lane >> 4) << 3;
  const int nc = lane & 15;
#pragma unroll
  for (int i = 0; i < 2; ++i)
#pragma unroll
    for (int j = 0; j < 2; ++j)
#pragma unroll
      for (int r = 0; r < 8; ++r) {
        const int Mr = m0 + i * 16 + r + hi;
        const int Nc = n0 + j * 16 + nc;
        const float h1 = acc1[i][j][r];
        const float h3 = acc3[i][j][r];
        const float sg = 1.f / (1.f + __expf(-h1));
        Gout[(size_t)Mr * N + Nc] = (_Float16)(h1 * sg * h3);
      }
}

// ---------------------------------------------------------------------------
// Flash attention, causal. One wave per (b, h, 16 query rows).
// Q,K: f16 [TOK, DM] (head slice at h*HD). Vt: f16 [b][h][HD][SEQ].
// Out: f16 [TOK, DM].
// ---------------------------------------------------------------------------
__global__ __launch_bounds__(32)
void k_attention(const _Float16* __restrict__ Q, const _Float16* __restrict__ Km,
                 const _Float16* __restrict__ Vt, _Float16* __restrict__ Out) {
  const int lane = threadIdx.x & 31;
  const int qblk = blockIdx.x;
  const int h    = blockIdx.y;
  const int b    = blockIdx.z;
  const int q0l  = qblk * 16;
  const int rowbase = b * SEQ;

  __shared__ __align__(16) _Float16 pbuf[16 * 32];

  const _Float16* Qb = Q + (size_t)(rowbase + q0l) * DM + h * HD;
  const v16h qa0 = load_a_frag(Qb, DM);
  const v16h qa1 = load_a_frag(Qb + 32, DM);
  const _Float16* Vb = Vt + (size_t)(b * NH + h) * HD * SEQ;

  v8f o[4] = {};
  float mrow[8], lrow[8];
#pragma unroll
  for (int r = 0; r < 8; ++r) { mrow[r] = -1e30f; lrow[r] = 0.f; }

  const int hi = (lane >> 4) << 3;
  const int nc = lane & 15;
  const float scale = 0.125f;

  const int nIter = (q0l + 16 + 31) >> 5;
  for (int it = 0; it < nIter; ++it) {
    const int ks = it * 32;
    v8f s0 = {}, s1 = {};
    {
      const _Float16* Kb = Km + (size_t)(rowbase + ks) * DM + h * HD;
      v16h b0 = load_b_frag(Kb, DM);
      v16h b1 = load_b_frag(Kb + 32, DM);
      s0 = wmma_f16(qa0, b0, s0);
      s0 = wmma_f16(qa1, b1, s0);
      v16h b2 = load_b_frag(Kb + (size_t)16 * DM, DM);
      v16h b3 = load_b_frag(Kb + (size_t)16 * DM + 32, DM);
      s1 = wmma_f16(qa0, b2, s1);
      s1 = wmma_f16(qa1, b3, s1);
    }
    __syncthreads();
#pragma unroll
    for (int r = 0; r < 8; ++r) {
      const int qrow = q0l + r + hi;
      float v0 = (ks + nc      <= qrow) ? s0[r] * scale : -1e30f;
      float v1 = (ks + 16 + nc <= qrow) ? s1[r] * scale : -1e30f;
      float mloc = fmaxf(v0, v1);
#pragma unroll
      for (int m = 8; m; m >>= 1) mloc = fmaxf(mloc, __shfl_xor(mloc, m, 32));
      const float mnew = fmaxf(mrow[r], mloc);
      const float corr = __expf(mrow[r] - mnew);
      const float p0 = __expf(v0 - mnew);
      const float p1 = __expf(v1 - mnew);
      float rs = p0 + p1;
#pragma unroll
      for (int m = 8; m; m >>= 1) rs += __shfl_xor(rs, m, 32);
      lrow[r] = lrow[r] * corr + rs;
      mrow[r] = mnew;
#pragma unroll
      for (int j = 0; j < 4; ++j) o[j][r] *= corr;
      const int ml = r + hi;
      pbuf[ml * 32 + nc]      = (_Float16)p0;
      pbuf[ml * 32 + 16 + nc] = (_Float16)p1;
    }
    __syncthreads();
    const v16h pa = load_a_frag(pbuf, 32);
#pragma unroll
    for (int j = 0; j < 4; ++j) {
      v16h vb = load_b_frag(Vb + (size_t)(j * 16) * SEQ + ks, SEQ);
      o[j] = wmma_f16(pa, vb, o[j]);
    }
  }

#pragma unroll
  for (int r = 0; r < 8; ++r) {
    const float inv = 1.f / lrow[r];
    const int Mr = rowbase + q0l + r + hi;
#pragma unroll
    for (int j = 0; j < 4; ++j)
      Out[(size_t)Mr * DM + h * HD + j * 16 + nc] = (_Float16)(o[j][r] * inv);
  }
}

// ---------------------------------------------------------------------------
// Host launcher
// ---------------------------------------------------------------------------
extern "C" void kernel_launch(void* const* d_in, const int* in_sizes, int n_in,
                              void* d_out, int out_size, void* d_ws, size_t ws_size,
                              hipStream_t stream) {
  (void)in_sizes; (void)n_in; (void)out_size; (void)ws_size;
  const float* x   = (const float*)d_in[0];
  const float* W_Q = (const float*)d_in[1];
  const float* W_K = (const float*)d_in[2];
  const float* W_V = (const float*)d_in[3];
  const float* W_O = (const float*)d_in[4];
  const float* g1  = (const float*)d_in[5];
  const float* g2  = (const float*)d_in[6];
  const float* w1  = (const float*)d_in[7];
  const float* w2  = (const float*)d_in[8];
  const float* w3  = (const float*)d_in[9];

  char* ws = (char*)d_ws;
  size_t off = 0;
  auto alloc = [&](size_t bytes) -> char* {
    char* p = ws + off;
    off += (bytes + 255) & ~(size_t)255;
    return p;
  };

  _Float16* WQh  = (_Float16*)alloc((size_t)DM * DM * 2);
  _Float16* WKh  = (_Float16*)alloc((size_t)DM * DM * 2);
  _Float16* WVh  = (_Float16*)alloc((size_t)DM * DM * 2);
  _Float16* WOh  = (_Float16*)alloc((size_t)DM * DM * 2);
  _Float16* w1h  = (_Float16*)alloc((size_t)DFF * DM * 2);
  _Float16* w2h  = (_Float16*)alloc((size_t)DM * DFF * 2);
  _Float16* w3h  = (_Float16*)alloc((size_t)DFF * DM * 2);
  _Float16* xn   = (_Float16*)alloc((size_t)TOK * DM * 2);
  _Float16* qh   = (_Float16*)alloc((size_t)TOK * DM * 2);
  _Float16* kh   = (_Float16*)alloc((size_t)TOK * DM * 2);
  _Float16* vt   = (_Float16*)alloc((size_t)TOK * DM * 2);
  _Float16* attn = (_Float16*)alloc((size_t)TOK * DM * 2);
  float*    y1   = (float*)   alloc((size_t)TOK * DM * 4);
  _Float16* xn2  = (_Float16*)alloc((size_t)TOK * DM * 2);
  _Float16* gate = (_Float16*)alloc((size_t)TOK * DFF * 2);

  const int nW = DM * DM, nF = DFF * DM;
  k_cast_f16<<<(nW + 255) / 256, 256, 0, stream>>>(W_Q, WQh, nW);
  k_cast_f16<<<(nW + 255) / 256, 256, 0, stream>>>(W_K, WKh, nW);
  k_cast_f16<<<(nW + 255) / 256, 256, 0, stream>>>(W_V, WVh, nW);
  k_cast_f16<<<(nW + 255) / 256, 256, 0, stream>>>(W_O, WOh, nW);
  k_cast_f16<<<(nF + 255) / 256, 256, 0, stream>>>(w1, w1h, nF);
  k_cast_f16<<<(nF + 255) / 256, 256, 0, stream>>>(w2, w2h, nF);
  k_cast_f16<<<(nF + 255) / 256, 256, 0, stream>>>(w3, w3h, nF);

  k_rmsnorm<<<TOK, 256, 0, stream>>>(x, g1, xn);

  dim3 gproj(TOK / 64, DM / 256);
  k_gemm<EPI_F16><<<gproj, 128, 0, stream>>>(xn, WQh, TOK, DM, DM, qh, nullptr);
  k_gemm<EPI_F16><<<gproj, 128, 0, stream>>>(xn, WKh, TOK, DM, DM, kh, nullptr);
  k_gemm<EPI_VT> <<<gproj, 128, 0, stream>>>(xn, WVh, TOK, DM, DM, vt, nullptr);

  dim3 gattn(SEQ / 16, NH, BATCH);
  k_attention<<<gattn, 32, 0, stream>>>(qh, kh, vt, attn);

  k_gemm<EPI_F32_RES><<<gproj, 128, 0, stream>>>(attn, WOh, TOK, DM, DM, y1, x);

  k_rmsnorm<<<TOK, 256, 0, stream>>>(y1, g2, xn2);

  dim3 ggate(TOK / 32, DFF / 128);
  k_gemm_gate<<<ggate, 128, 0, stream>>>(xn2, w1h, w3h, TOK, DFF, DM, gate);

  k_gemm<EPI_F32_RES><<<gproj, 128, 0, stream>>>(gate, w2h, TOK, DM, DFF,
                                                 (float*)d_out, y1);
}
